// RegionProposalNetwork_1606317769511
// MI455X (gfx1250) — matrix operand order repaired and checked by reading, not compiled, not run on hardware
//
#include <hip/hip_runtime.h>

// ---------------- types for WMMA ----------------
typedef __attribute__((ext_vector_type(16))) __bf16 v16bf;
typedef __attribute__((ext_vector_type(8)))  float  v8f;
typedef __attribute__((ext_vector_type(4)))  unsigned int v4u;

union Frag {
  v16bf v;
  v4u   q[2];
};

__device__ inline unsigned short f2bf(float f) {
  unsigned u = __float_as_uint(f);
  u += 0x7FFFu + ((u >> 16) & 1u);   // round-to-nearest-even
  return (unsigned short)(u >> 16);
}
__device__ inline float bf2f(unsigned short s) {
  return __uint_as_float(((unsigned)s) << 16);
}
__device__ inline v8f wmma_bf16(const Frag& a, const Frag& b, v8f c) {
  return __builtin_amdgcn_wmma_f32_16x16x32_bf16(false, a.v, false, b.v,
                                                 (short)0, c, false, false);
}

// ---------------- problem constants ----------------
constexpr int NBATCH = 2;
constexpr int HH = 100, WWI = 100, HWP = HH * WWI;   // 10000 pixels
constexpr int PH = 102, PW = 102;                    // zero-padded input plane
constexpr int CIN = 256, CMID = 256;
constexpr int KT3 = 72;          // 2304 / 32 K-tiles for 3x3 conv
constexpr int NT3 = 16;          // 256/16 output-channel tiles
constexpr int MTI = 625;         // 10000/16 pixel tiles
constexpr int NT1 = 4, KT1 = 8;  // 1x1 conv: 64 combined channels, K=256
constexpr int NANCH = 9;
constexpr int NA = HWP * NANCH;  // 90000 anchors
constexpr int PRE = 6000, POST = 300;
constexpr int SORTN = 131072;    // pow2 >= 90000
constexpr int SORTLG = 17;
constexpr int MWORDS = (PRE + 31) / 32;  // 188

// interleaved packed-B strides (elements)
constexpr int WB3_KT = 32 * NT3 * 16;   // 8192 elems (16KB) per K-tile
constexpr int WB1_KT = 32 * NT1 * 16;   // 2048 elems (4KB) per K-tile

// d_out element offsets (all float32-sized elements)
constexpr size_t O_LOCS   = 0;
constexpr size_t O_SCORES = (size_t)NBATCH * NA * 4;            // 720000
constexpr size_t O_ROIS   = O_SCORES + (size_t)NBATCH * NA * 2; // 1080000
constexpr size_t O_RIDX   = O_ROIS + (size_t)NBATCH * POST * 4; // 1082400
constexpr size_t O_ANCH   = O_RIDX + (size_t)NBATCH * POST;     // 1083000

// workspace byte offsets
constexpr size_t ALN(size_t x) { return (x + 255) & ~(size_t)255; }
constexpr size_t SZ_XP  = (size_t)NBATCH * PH * PW * CIN * 2;   // padded x plane (ushort)
constexpr size_t SZ_H   = (size_t)NBATCH * HWP * CMID * 2;      // h plane (ushort)
constexpr size_t SZ_WB  = (size_t)KT3 * WB3_KT * 2;
constexpr size_t SZ_W1B = (size_t)KT1 * WB1_KT * 2;
constexpr size_t W_XHI   = 0;
constexpr size_t W_XLO   = ALN(W_XHI + SZ_XP);
constexpr size_t W_HHI   = ALN(W_XLO + SZ_XP);
constexpr size_t W_HLO   = ALN(W_HHI + SZ_H);
constexpr size_t W_WBHI  = ALN(W_HLO + SZ_H);
constexpr size_t W_WBLO  = ALN(W_WBHI + SZ_WB);
constexpr size_t W_W1BHI = ALN(W_WBLO + SZ_WB);
constexpr size_t W_W1BLO = ALN(W_W1BHI + SZ_W1B);
constexpr size_t W_BOXES = ALN(W_W1BLO + SZ_W1B);
constexpr size_t W_SC    = ALN(W_BOXES + (size_t)NBATCH * NA * 4 * 4);
constexpr size_t W_KEYS  = ALN(W_SC + (size_t)NBATCH * NA * 4);
constexpr size_t W_TOPB  = ALN(W_KEYS + (size_t)NBATCH * SORTN * 8);
constexpr size_t W_TOPV  = ALN(W_TOPB + (size_t)NBATCH * PRE * 4 * 4);
constexpr size_t W_MASK  = ALN(W_TOPV + (size_t)NBATCH * PRE * 4);

// ---------------- zero-fill padded x planes ----------------
__global__ __launch_bounds__(256) void zero_pads(unsigned short* __restrict__ xhi,
                                                 unsigned short* __restrict__ xlo) {
  int tid = blockIdx.x * blockDim.x + threadIdx.x;
  if (tid >= NBATCH * PH * PW * CIN) return;
  xhi[tid] = 0;
  xlo[tid] = 0;
}

// ---------------- prep: x (NCHW f32) -> padded NHWC bf16 hi/lo planes ----------------
__global__ __launch_bounds__(256) void prep_x(const float* __restrict__ x,
                                              unsigned short* __restrict__ xhi,
                                              unsigned short* __restrict__ xlo) {
  int tid = blockIdx.x * blockDim.x + threadIdx.x;
  if (tid >= NBATCH * HWP * CIN) return;
  int ci = tid & 255;
  int rest = tid >> 8;
  int p = rest % HWP;
  int n = rest / HWP;
  int y = p / WWI, xq = p - y * WWI;
  float v = x[((size_t)n * CIN + ci) * HWP + p];
  unsigned short h = f2bf(v);
  size_t dst = (((size_t)n * PH + (y + 1)) * PW + (xq + 1)) * CIN + ci;
  xhi[dst] = h;
  xlo[dst] = f2bf(v - bf2f(h));
}

// ---------------- prep: pack 3x3 weights, interleaved B-fragment layout ----------------
// wB[((kt*32 + lane)*16 + nt)*16 + e] ; col = nt*16+(lane&15), K = kt*32 + (e>=8?16:0) + (lane>=16?8:0) + (e&7)
// K ordering: K = (ky*3+kx)*256 + ci
__global__ __launch_bounds__(256) void prep_wb(const float* __restrict__ w,
                                               unsigned short* __restrict__ wbhi,
                                               unsigned short* __restrict__ wblo) {
  int t = blockIdx.x * blockDim.x + threadIdx.x;
  if (t >= KT3 * WB3_KT) return;
  int e = t & 15;
  int nt = (t >> 4) & 15;
  int l = (t >> 8) & 31;
  int kt = t >> 13;
  int co = nt * 16 + (l & 15);
  int K = kt * 32 + ((e >> 3) << 4) + ((l >> 4) << 3) + (e & 7);
  int kyx = K >> 8, ci = K & 255;
  int ky = kyx / 3, kx = kyx - ky * 3;
  float wv = w[(((size_t)co * CIN + ci) * 3 + ky) * 3 + kx];
  unsigned short h = f2bf(wv);
  wbhi[t] = h;
  wblo[t] = f2bf(wv - bf2f(h));
}

// ---------------- prep: pack combined 1x1 weights (36 loc + 18 score + pad) ----------------
// w1B[((kt*32 + lane)*4 + nt)*16 + e]
__global__ __launch_bounds__(256) void prep_w1b(const float* __restrict__ locw,
                                                const float* __restrict__ scw,
                                                unsigned short* __restrict__ whi,
                                                unsigned short* __restrict__ wlo) {
  int t = blockIdx.x * blockDim.x + threadIdx.x;
  if (t >= KT1 * WB1_KT) return;
  int e = t & 15;
  int nt = (t >> 4) & 3;
  int l = (t >> 6) & 31;
  int kt = t >> 11;
  int c = nt * 16 + (l & 15);
  int ci = kt * 32 + ((e >> 3) << 4) + ((l >> 4) << 3) + (e & 7);
  float wv = 0.f;
  if (c < 36) wv = locw[(size_t)c * CMID + ci];
  else if (c < 54) wv = scw[(size_t)(c - 36) * CMID + ci];
  unsigned short h = f2bf(wv);
  whi[t] = h;
  wlo[t] = f2bf(wv - bf2f(h));
}

// ---------------- 3x3 conv as implicit GEMM: one wave -> 16 pixels x 64 channels ----------------
// bf16x2 precision split: acc += a_lo*b_hi + a_hi*b_lo + a_hi*b_hi (fp32 accumulate)
__global__ __launch_bounds__(32) void conv3_wmma(
    const unsigned short* __restrict__ xhi, const unsigned short* __restrict__ xlo,
    const unsigned short* __restrict__ wbhi, const unsigned short* __restrict__ wblo,
    const float* __restrict__ bias,
    unsigned short* __restrict__ hhi, unsigned short* __restrict__ hlo) {
  const int lane = threadIdx.x;
  const int mt = blockIdx.x, ng = blockIdx.y, n = blockIdx.z;  // ng: group of 4 N-tiles
  const int row = lane & 15;
  const int hiHalf = lane >> 4;
  const int sA = hiHalf * 8;
  const int p = mt * 16 + row;
  const int y = p / WWI, x = p - y * WWI;
  const int nt0 = ng * 4;
  // per-wave packed-B base: all fragments at base + kt*WB3_KT + j*16 (immediate offsets)
  const int wb0 = (lane * NT3 + nt0) * 16;
  const unsigned short* whB = wbhi + wb0;
  const unsigned short* wlB = wblo + wb0;
  v8f acc[4] = {{}, {}, {}, {}};
  for (int kyx = 0; kyx < 9; ++kyx) {
    const int ky = kyx / 3, kx = kyx - ky * 3;
    // padded, branch-free A base for this filter tap
    const int abase = ((n * PH + (y + ky)) * PW + (x + kx)) * CIN + sA;
    const unsigned short* axh = xhi + abase;
    const unsigned short* axl = xlo + abase;
    const unsigned short* wh = whB + kyx * 8 * WB3_KT;
    const unsigned short* wl = wlB + kyx * 8 * WB3_KT;
#pragma unroll 4
    for (int c8 = 0; c8 < 8; ++c8) {
      Frag ah, al;
      ah.q[0] = *(const v4u*)(axh + c8 * 32);
      ah.q[1] = *(const v4u*)(axh + c8 * 32 + 16);
      al.q[0] = *(const v4u*)(axl + c8 * 32);
      al.q[1] = *(const v4u*)(axl + c8 * 32 + 16);
      Frag bh[4], bl[4];
#pragma unroll
      for (int j = 0; j < 4; ++j) {
        const int wo = c8 * WB3_KT + j * 16;
        bh[j].q[0] = *(const v4u*)(wh + wo);
        bh[j].q[1] = *(const v4u*)(wh + wo + 8);
        bl[j].q[0] = *(const v4u*)(wl + wo);
        bl[j].q[1] = *(const v4u*)(wl + wo + 8);
      }
      // term-major: same-accumulator WMMAs are 4 apart
#pragma unroll
      for (int j = 0; j < 4; ++j) acc[j] = wmma_bf16(al, bh[j], acc[j]);
#pragma unroll
      for (int j = 0; j < 4; ++j) acc[j] = wmma_bf16(ah, bl[j], acc[j]);
#pragma unroll
      for (int j = 0; j < 4; ++j) acc[j] = wmma_bf16(ah, bh[j], acc[j]);
    }
  }
#pragma unroll
  for (int j = 0; j < 4; ++j) {
    const int co = (nt0 + j) * 16 + row;
    const float bb = bias[co];
#pragma unroll
    for (int r = 0; r < 8; ++r) {
      const int pm = mt * 16 + r + hiHalf * 8;
      float v = acc[j][r] + bb;
      v = v > 0.f ? v : 0.f;   // ReLU
      unsigned short h = f2bf(v);
      const int o = (n * HWP + pm) * CMID + co;
      hhi[o] = h;
      hlo[o] = f2bf(v - bf2f(h));
    }
  }
}

// ---------------- 1x1 loc+score conv: one wave -> 16 pixels x 64 combined channels ----------------
__global__ __launch_bounds__(32) void conv1_wmma(
    const unsigned short* __restrict__ hhi, const unsigned short* __restrict__ hlo,
    const unsigned short* __restrict__ whi, const unsigned short* __restrict__ wlo,
    const float* __restrict__ locb, const float* __restrict__ scb,
    float* __restrict__ out) {
  const int lane = threadIdx.x;
  const int mt = blockIdx.x, n = blockIdx.z;
  const int row = lane & 15;
  const int hiHalf = lane >> 4;
  const int sA = hiHalf * 8;
  const int p = mt * 16 + row;
  const int abase = (n * HWP + p) * CMID + sA;
  const unsigned short* axh = hhi + abase;
  const unsigned short* axl = hlo + abase;
  const unsigned short* wh = whi + lane * (NT1 * 16);
  const unsigned short* wl = wlo + lane * (NT1 * 16);
  v8f acc[4] = {{}, {}, {}, {}};
#pragma unroll 4
  for (int kt = 0; kt < KT1; ++kt) {
    Frag ah, al;
    ah.q[0] = *(const v4u*)(axh + kt * 32);
    ah.q[1] = *(const v4u*)(axh + kt * 32 + 16);
    al.q[0] = *(const v4u*)(axl + kt * 32);
    al.q[1] = *(const v4u*)(axl + kt * 32 + 16);
    Frag bh[4], bl[4];
#pragma unroll
    for (int j = 0; j < 4; ++j) {
      const int wo = kt * WB1_KT + j * 16;
      bh[j].q[0] = *(const v4u*)(wh + wo);
      bh[j].q[1] = *(const v4u*)(wh + wo + 8);
      bl[j].q[0] = *(const v4u*)(wl + wo);
      bl[j].q[1] = *(const v4u*)(wl + wo + 8);
    }
#pragma unroll
    for (int j = 0; j < 4; ++j) acc[j] = wmma_bf16(al, bh[j], acc[j]);
#pragma unroll
    for (int j = 0; j < 4; ++j) acc[j] = wmma_bf16(ah, bl[j], acc[j]);
#pragma unroll
    for (int j = 0; j < 4; ++j) acc[j] = wmma_bf16(ah, bh[j], acc[j]);
  }
#pragma unroll
  for (int j = 0; j < 4; ++j) {
    const int c = j * 16 + row;
#pragma unroll
    for (int r = 0; r < 8; ++r) {
      const int pm = mt * 16 + r + hiHalf * 8;
      float v = acc[j][r];
      if (c < 36)
        out[O_LOCS + (size_t)n * NA * 4 + (size_t)pm * 36 + c] = v + locb[c];
      else if (c < 54)
        out[O_SCORES + (size_t)n * NA * 2 + (size_t)pm * 18 + (c - 36)] = v + scb[c - 36];
    }
  }
}

// ---------------- anchors ----------------
__global__ __launch_bounds__(256) void anchors_kernel(float* __restrict__ out) {
  int i = blockIdx.x * blockDim.x + threadIdx.x;
  if (i >= NA) return;
  int p = i / NANCH, a = i - p * NANCH;
  int rI = a / 3, sI = a - rI * 3;
  const float ratios[3] = {0.5f, 1.0f, 2.0f};
  const float scales[3] = {8.0f, 16.0f, 32.0f};
  float h = 16.f * scales[sI] * sqrtf(ratios[rI]);
  float w = 16.f * scales[sI] * sqrtf(1.0f / ratios[rI]);
  int y = p / WWI, x = p - y * WWI;
  float sy = y * 16.f, sx = x * 16.f;
  float* A = out + O_ANCH + (size_t)i * 4;
  A[0] = sy + 8.f - 0.5f * h;
  A[1] = sx + 8.f - 0.5f * w;
  A[2] = sy + 8.f + 0.5f * h;
  A[3] = sx + 8.f + 0.5f * w;
}

// ---------------- proposals: loc2bbox + clip + validity + sort keys ----------------
__global__ __launch_bounds__(256) void boxes_keys(const float* __restrict__ out,
                                                  const int* __restrict__ imh,
                                                  const int* __restrict__ imw,
                                                  float* __restrict__ boxes,
                                                  float* __restrict__ scAll,
                                                  unsigned long long* __restrict__ keys) {
  int tid = blockIdx.x * blockDim.x + threadIdx.x;
  if (tid >= NBATCH * SORTN) return;
  int n = tid >> SORTLG;
  int i = tid & (SORTN - 1);
  unsigned long long key;
  if (i < NA) {
    const float* A = out + O_ANCH + (size_t)i * 4;
    float a0 = A[0], a1 = A[1], a2 = A[2], a3 = A[3];
    float h = a2 - a0, w = a3 - a1;
    float cy = a0 + 0.5f * h, cx = a1 + 0.5f * w;
    const float* L = out + O_LOCS + (size_t)n * NA * 4 + (size_t)i * 4;
    float dy = L[0], dx = L[1], dh = L[2], dw = L[3];
    cy += dy * h;
    cx += dx * w;
    h *= expf(dh);
    w *= expf(dw);
    float fh = (float)imh[0], fw = (float)imw[0];
    float y0 = fminf(fmaxf(cy - 0.5f * h, 0.f), fh);
    float x0 = fminf(fmaxf(cx - 0.5f * w, 0.f), fw);
    float y1 = fminf(fmaxf(cy + 0.5f * h, 0.f), fh);
    float x1 = fminf(fmaxf(cx + 0.5f * w, 0.f), fw);
    float* B = boxes + ((size_t)n * NA + i) * 4;
    B[0] = y0; B[1] = x0; B[2] = y1; B[3] = x1;
    bool valid = (y1 - y0 >= 16.f) && (x1 - x0 >= 16.f);
    float fg = out[O_SCORES + (size_t)n * NA * 2 + (size_t)i * 2 + 1];
    float negInf = __uint_as_float(0xFF800000u);
    float sc = valid ? fg : negInf;
    scAll[(size_t)n * NA + i] = sc;
    unsigned u = __float_as_uint(sc);
    unsigned m = (u >> 31) ? ~u : (u | 0x80000000u);  // monotone map
    key = (((unsigned long long)(~m)) << 32) | (unsigned)i; // asc key = desc score, tie: low idx
  } else {
    key = (0xFFFFFFFFull << 32) | (unsigned)i;  // padding sorts last
  }
  keys[(size_t)n * SORTN + i] = key;
}

// ---------------- global bitonic sort step ----------------
__global__ __launch_bounds__(256) void bitonic_step(unsigned long long* __restrict__ keys,
                                                    int j, int k) {
  int tid = blockIdx.x * blockDim.x + threadIdx.x;
  if (tid >= NBATCH * SORTN) return;
  int n = tid >> SORTLG;
  int i = tid & (SORTN - 1);
  int ij = i ^ j;
  if (ij > i) {
    unsigned long long* a = keys + ((size_t)n << SORTLG);
    unsigned long long A = a[i], B = a[ij];
    bool asc = ((i & k) == 0);
    if ((A > B) == asc) { a[i] = B; a[ij] = A; }
  }
}

// ---------------- gather top PRE boxes ----------------
__global__ __launch_bounds__(256) void gather_top(const unsigned long long* __restrict__ keys,
                                                  const float* __restrict__ boxes,
                                                  const float* __restrict__ scAll,
                                                  float* __restrict__ topB,
                                                  int* __restrict__ topV) {
  int tid = blockIdx.x * blockDim.x + threadIdx.x;
  if (tid >= NBATCH * PRE) return;
  int n = tid / PRE, t = tid - n * PRE;
  unsigned long long key = keys[(size_t)n * SORTN + t];
  unsigned idx = (unsigned)(key & 0xFFFFFFFFu);
  float* T = topB + ((size_t)n * PRE + t) * 4;
  if (idx < (unsigned)NA) {
    const float* B = boxes + ((size_t)n * NA + idx) * 4;
    T[0] = B[0]; T[1] = B[1]; T[2] = B[2]; T[3] = B[3];
    float negInf = __uint_as_float(0xFF800000u);
    topV[(size_t)n * PRE + t] = (scAll[(size_t)n * NA + idx] > negInf) ? 1 : 0;
  } else {
    T[0] = T[1] = T[2] = T[3] = 0.f;
    topV[(size_t)n * PRE + t] = 0;
  }
}

// ---------------- NMS suppression bitmask: one thread per (row, 32-col word) ----------------
__global__ __launch_bounds__(256) void nms_mask(const float* __restrict__ topB,
                                                unsigned* __restrict__ mask) {
  int tid = blockIdx.x * blockDim.x + threadIdx.x;
  if (tid >= NBATCH * PRE * MWORDS) return;
  int w = tid % MWORDS;
  int rest = tid / MWORDS;
  int r = rest % PRE;
  int n = rest / PRE;
  const float* B = topB + (size_t)n * PRE * 4;
  float ry0 = B[r * 4 + 0], rx0 = B[r * 4 + 1], ry1 = B[r * 4 + 2], rx1 = B[r * 4 + 3];
  float rArea = (ry1 - ry0) * (rx1 - rx0);
  unsigned m = 0;
  for (int bit = 0; bit < 32; ++bit) {
    int c = w * 32 + bit;
    if (c < PRE && c > r) {
      float cy0 = B[c * 4 + 0], cx0 = B[c * 4 + 1], cy1 = B[c * 4 + 2], cx1 = B[c * 4 + 3];
      float cArea = (cy1 - cy0) * (cx1 - cx0);
      float ty = fmaxf(ry0, cy0), tx = fmaxf(rx0, cx0);
      float by = fminf(ry1, cy1), bx = fminf(rx1, cx1);
      float ih = fmaxf(by - ty, 0.f), iw = fmaxf(bx - tx, 0.f);
      float inter = ih * iw;
      float iou = inter / (rArea + cArea - inter + 1e-9f);
      if (iou > 0.7f) m |= (1u << bit);
    }
  }
  mask[tid] = m;
}

// ---------------- sequential NMS reduce + emit rois / roi_indices ----------------
__global__ __launch_bounds__(256) void nms_reduce(const float* __restrict__ topB,
                                                  const int* __restrict__ topV,
                                                  const unsigned* __restrict__ mask,
                                                  float* __restrict__ out) {
  const int n = blockIdx.x;
  const int tid = threadIdx.x;
  __shared__ unsigned removed[MWORDS];
  __shared__ int flag;
  for (int t = tid; t < MWORDS; t += 256) removed[t] = 0;
  __syncthreads();
  for (int i = 0; i < PRE; ++i) {
    if (tid == 0)
      flag = topV[(size_t)n * PRE + i] && !((removed[i >> 5] >> (i & 31)) & 1u);
    __syncthreads();
    if (flag) {
      for (int t = tid; t < MWORDS; t += 256)
        removed[t] |= mask[((size_t)n * PRE + i) * MWORDS + t];
    }
    __syncthreads();
  }
  if (tid == 0) {
    int cnt = 0;
    for (int i = 0; i < PRE && cnt < POST; ++i) {
      bool keep = topV[(size_t)n * PRE + i] && !((removed[i >> 5] >> (i & 31)) & 1u);
      if (keep) {
        const float* B = topB + ((size_t)n * PRE + i) * 4;
        float* R = out + O_ROIS + ((size_t)n * POST + cnt) * 4;
        R[0] = B[0]; R[1] = B[1]; R[2] = B[2]; R[3] = B[3];
        ++cnt;
      }
    }
    for (; cnt < POST; ++cnt) {
      float* R = out + O_ROIS + ((size_t)n * POST + cnt) * 4;
      R[0] = R[1] = R[2] = R[3] = 0.f;
    }
  }
  int* oi = (int*)out;
  for (int t = tid; t < POST; t += 256)
    oi[O_RIDX + (size_t)n * POST + t] = n;
}

// ---------------- host launcher ----------------
extern "C" void kernel_launch(void* const* d_in, const int* in_sizes, int n_in,
                              void* d_out, int out_size, void* d_ws, size_t ws_size,
                              hipStream_t stream) {
  const float* x    = (const float*)d_in[0];
  const float* w1   = (const float*)d_in[1];
  const float* b1   = (const float*)d_in[2];
  const float* locw = (const float*)d_in[3];
  const float* locb = (const float*)d_in[4];
  const float* scw  = (const float*)d_in[5];
  const float* scb  = (const float*)d_in[6];
  const int*   imh  = (const int*)d_in[7];
  const int*   imw  = (const int*)d_in[8];
  float* out = (float*)d_out;
  char* ws = (char*)d_ws;

  unsigned short* xhi   = (unsigned short*)(ws + W_XHI);
  unsigned short* xlo   = (unsigned short*)(ws + W_XLO);
  unsigned short* hhi   = (unsigned short*)(ws + W_HHI);
  unsigned short* hlo   = (unsigned short*)(ws + W_HLO);
  unsigned short* wbhi  = (unsigned short*)(ws + W_WBHI);
  unsigned short* wblo  = (unsigned short*)(ws + W_WBLO);
  unsigned short* w1bhi = (unsigned short*)(ws + W_W1BHI);
  unsigned short* w1blo = (unsigned short*)(ws + W_W1BLO);
  float* boxes = (float*)(ws + W_BOXES);
  float* scAll = (float*)(ws + W_SC);
  unsigned long long* keys = (unsigned long long*)(ws + W_KEYS);
  float* topB = (float*)(ws + W_TOPB);
  int*   topV = (int*)(ws + W_TOPV);
  unsigned* mask = (unsigned*)(ws + W_MASK);

  // 1) precision-split / layout transforms (padded x => branch-free conv loop)
  zero_pads<<<(NBATCH * PH * PW * CIN + 255) / 256, 256, 0, stream>>>(xhi, xlo);
  prep_x<<<(NBATCH * HWP * CIN + 255) / 256, 256, 0, stream>>>(x, xhi, xlo);
  prep_wb<<<(KT3 * WB3_KT + 255) / 256, 256, 0, stream>>>(w1, wbhi, wblo);
  prep_w1b<<<(KT1 * WB1_KT + 255) / 256, 256, 0, stream>>>(locw, scw, w1bhi, w1blo);

  // 2) 3x3 conv + ReLU via WMMA (bf16x2 split, 16x64 tile per wave)
  conv3_wmma<<<dim3(MTI, NT3 / 4, NBATCH), 32, 0, stream>>>(xhi, xlo, wbhi, wblo, b1, hhi, hlo);

  // 3) 1x1 loc+score convs via WMMA -> rpn_locs, rpn_scores in d_out
  conv1_wmma<<<dim3(MTI, 1, NBATCH), 32, 0, stream>>>(hhi, hlo, w1bhi, w1blo, locb, scb, out);

  // 4) anchors
  anchors_kernel<<<(NA + 255) / 256, 256, 0, stream>>>(out);

  // 5) proposals + sort keys
  boxes_keys<<<(NBATCH * SORTN + 255) / 256, 256, 0, stream>>>(out, imh, imw, boxes, scAll, keys);

  // 6) bitonic argsort (descending score, index tie-break)
  for (int k = 2; k <= SORTN; k <<= 1)
    for (int j = k >> 1; j > 0; j >>= 1)
      bitonic_step<<<(NBATCH * SORTN + 255) / 256, 256, 0, stream>>>(keys, j, k);

  // 7) top-6000 gather, NMS mask, sequential reduce -> rois + roi_indices
  gather_top<<<(NBATCH * PRE + 255) / 256, 256, 0, stream>>>(keys, boxes, scAll, topB, topV);
  nms_mask<<<(NBATCH * PRE * MWORDS + 255) / 256, 256, 0, stream>>>(topB, mask);
  nms_reduce<<<NBATCH, 256, 0, stream>>>(topB, topV, mask, out);
}
